// TransformerBlockTorch_77455440216458
// MI455X (gfx1250) — matrix operand order repaired and checked by reading, not compiled, not run on hardware
//
#include <hip/hip_runtime.h>

// ---------------------------------------------------------------------------
// Transformer block for MI455X (gfx1250, wave32, WMMA + TDM).
// GEMMs: v_wmma_f32_16x16x32_bf16, LDS tiles staged by tensor_load_to_lds
// (double-buffered, s_wait_tensorcnt pipelined).  bf16 weights live in L2.
// ---------------------------------------------------------------------------

typedef __attribute__((ext_vector_type(8)))  float  v8f;
typedef __attribute__((ext_vector_type(8)))  __bf16 v8bf;
typedef __attribute__((ext_vector_type(16))) __bf16 v16bf;
typedef __attribute__((ext_vector_type(4)))  unsigned int u32x4;
typedef __attribute__((ext_vector_type(4)))  int i32x4;
typedef __attribute__((ext_vector_type(8)))  int i32x8;

#define DEV __device__ __forceinline__

constexpr int Bc  = 2;
constexpr int Sc  = 2048;
constexpr int Hc  = 2048;
constexpr int NHc = 16;
constexpr int HDc = 128;
constexpr int FFc = 8192;

// ---- TDM availability probe (ROCm 7.2 = 5-arg builtin; therock = 6-arg) ----
#if defined(__HIP_DEVICE_COMPILE__) && defined(__has_builtin)
#if __has_builtin(__builtin_amdgcn_tensor_load_to_lds) && \
    __has_builtin(__builtin_amdgcn_s_wait_tensorcnt)
#define HAVE_TDM 1
#if defined(__has_include)
#if __has_include(<hip/amd_detail/amd_gfx1250_TDM.h>)
#define TDM_6ARG 1
#else
#define TDM_6ARG 0
#endif
#else
#define TDM_6ARG 0
#endif
#endif
#endif
#ifndef HAVE_TDM
#define HAVE_TDM 0
#endif

DEV unsigned short f2bf(float f) {
    unsigned int u = __float_as_uint(f);
    u += 0x7fffu + ((u >> 16) & 1u);   // round-to-nearest-even
    return (unsigned short)(u >> 16);
}

DEV v16bf cat8(v8bf lo, v8bf hi) {
    return __builtin_shufflevector(lo, hi, 0,1,2,3,4,5,6,7,8,9,10,11,12,13,14,15);
}

#if HAVE_TDM
// ---------------------------------------------------------------------------
// TDM: DMA a 128-row x 32-col bf16 tile (row stride K elements) from global
// into LDS at lds_off, inserting 4 pad DWORDs after every 16 data DWORDs
// (row stride in LDS = 32 data + 8 pad bf16 = 40, matching fragment reads).
// D# per CDNA5 ISA ch.8: group0 {count,lds,global,type=2}, group1 dims/strides.
// ---------------------------------------------------------------------------
DEV void tdm_load_tile(unsigned lds_off, const unsigned short* gptr, int K) {
    unsigned long long ga = (unsigned long long)(uintptr_t)gptr;
    u32x4 g0;
    g0[0] = 1u;                                   // count=1, user mode
    g0[1] = lds_off;                              // lds_addr (bytes)
    g0[2] = (unsigned)ga;                         // global_addr[31:0]
    g0[3] = (unsigned)(ga >> 32) | (2u << 30);    // global_addr[56:32] | type=2
    i32x8 g1;
    g1[0] = (1 << 16)      // data_size = 2 bytes
          | (1 << 20)      // pad_enable
          | (3 << 22)      // pad_interval: 16 DWORDs
          | (3 << 25);     // pad_amount:   4 DWORDs
    g1[1] = (int)(32u << 16);      // tensor_dim0 = 32 (tile always in-bounds)
    g1[2] = (int)(128u << 16);     // tensor_dim1 = 128
    g1[3] = (int)(32u << 16);      // tile_dim0 = 32
    g1[4] = 128;                   // tile_dim1 = 128, tile_dim2 = 0
    g1[5] = K;                     // tensor_dim0_stride (elements)
    g1[6] = 0;
    g1[7] = 0;
    i32x4 gz = {0, 0, 0, 0};
#if TDM_6ARG
    i32x8 z8 = {0, 0, 0, 0, 0, 0, 0, 0};
    __builtin_amdgcn_tensor_load_to_lds(g0, g1, gz, gz, z8, 0);
#else
    __builtin_amdgcn_tensor_load_to_lds(g0, g1, gz, gz, 0);
#endif
}
#endif // HAVE_TDM

// ---------------------------------------------------------------------------
// LayerNorm (f32 in) -> bf16 out.  One block (256 thr) per row of H=2048.
// ---------------------------------------------------------------------------
__global__ void ln_bf16_kernel(const float* __restrict__ x,
                               const float* __restrict__ g,
                               const float* __restrict__ b,
                               unsigned short* __restrict__ out) {
    const int row = blockIdx.x;
    const float* xr = x + (size_t)row * Hc;
    unsigned short* orow = out + (size_t)row * Hc;

    float s = 0.f, s2 = 0.f;
    for (int i = threadIdx.x; i < Hc; i += 256) {
        float v = xr[i];
        s += v; s2 += v * v;
    }
    for (int off = 16; off >= 1; off >>= 1) {
        s  += __shfl_xor(s,  off);
        s2 += __shfl_xor(s2, off);
    }
    __shared__ float sh[16];
    const int wid = threadIdx.x >> 5, ln = threadIdx.x & 31;
    if (ln == 0) { sh[wid] = s; sh[wid + 8] = s2; }
    __syncthreads();
    if (threadIdx.x == 0) {
        float ts = 0.f, ts2 = 0.f;
        for (int w = 0; w < 8; w++) { ts += sh[w]; ts2 += sh[w + 8]; }
        float mu  = ts  * (1.f / Hc);
        float var = ts2 * (1.f / Hc) - mu * mu;
        sh[0] = mu;
        sh[1] = rsqrtf(var + 1e-5f);
    }
    __syncthreads();
    const float mu = sh[0], rstd = sh[1];
    for (int i = threadIdx.x; i < Hc; i += 256)
        orow[i] = f2bf((xr[i] - mu) * rstd * g[i] + b[i]);
}

// ---------------------------------------------------------------------------
// Weight cast+transpose: W (K x N, f32) -> Wt (N x K, bf16).  32x32 LDS tile.
// ---------------------------------------------------------------------------
__global__ void wcast_t_kernel(const float* __restrict__ W,
                               unsigned short* __restrict__ Wt,
                               int K, int N) {
    __shared__ float tile[32][33];
    const int tx = threadIdx.x, ty = threadIdx.y;
    const int n0 = blockIdx.x * 32, k0 = blockIdx.y * 32;
#pragma unroll
    for (int j = 0; j < 4; j++)
        tile[ty + j * 8][tx] = W[(size_t)(k0 + ty + j * 8) * N + n0 + tx];
    __syncthreads();
#pragma unroll
    for (int j = 0; j < 4; j++)
        Wt[(size_t)(n0 + ty + j * 8) * K + k0 + tx] = f2bf(tile[tx][ty + j * 8]);
}

// ---------------------------------------------------------------------------
// Per-head V transpose: v (B,S,NH*HD) bf16 -> vt (B,NH,HD,S) bf16.
// ---------------------------------------------------------------------------
__global__ void vtrans_kernel(const unsigned short* __restrict__ v,
                              unsigned short* __restrict__ vt) {
    __shared__ unsigned short tile[32][33];
    const int tx = threadIdx.x, ty = threadIdx.y;
    const int bh = blockIdx.z;                      // b*NH + h
    const int s0 = blockIdx.x * 32, d0 = blockIdx.y * 32;
    const unsigned short* src = v + (size_t)(bh / NHc) * Sc * Hc + (size_t)(bh % NHc) * HDc;
    unsigned short* dst = vt + (size_t)bh * HDc * Sc;
#pragma unroll
    for (int j = 0; j < 4; j++)
        tile[ty + j * 8][tx] = src[(size_t)(s0 + ty + j * 8) * Hc + d0 + tx];
    __syncthreads();
#pragma unroll
    for (int j = 0; j < 4; j++)
        dst[(size_t)(d0 + ty + j * 8) * Sc + s0 + tx] = tile[tx][ty + j * 8];
}

// ---------------------------------------------------------------------------
// WMMA GEMM:  C(MxN, f32 acc) = A(MxK, bf16 row-major) * Bt(NxK, bf16)^T
// Block tile 128x128x32, 256 threads = 8 waves (2x4), wave tile 64x32.
// TDM double-buffered staging (wave 0 drives the DMA, s_wait_tensorcnt
// pipelined: issue tile t+1, wait <=2 => tile t resident).
// mode 0: store bf16;  mode 1: gelu(tanh) -> bf16;  mode 2: f32 = R + acc.
// ---------------------------------------------------------------------------
__global__ void __launch_bounds__(256)
gemm_bf16_kernel(const unsigned short* __restrict__ A,
                 const unsigned short* __restrict__ Bt,
                 unsigned short* __restrict__ Cb,
                 float* Cf, const float* R,
                 int M, int N, int K, int mode) {
    (void)M;
    __shared__ __align__(16) unsigned short As[2][128 * 40];
    __shared__ __align__(16) unsigned short Bs[2][128 * 40];

    const int tid   = threadIdx.x;
    const int lane  = tid & 31, wave = tid >> 5;
    const int waveM = wave & 1, waveN = wave >> 1;
    const int rowBase = blockIdx.y * 128, colBase = blockIdx.x * 128;
    const int m16 = lane & 15, hseg = lane >> 4;

    const v8f vzero = {0.f,0.f,0.f,0.f,0.f,0.f,0.f,0.f};
    v8f acc[4][2];
#pragma unroll
    for (int mt = 0; mt < 4; mt++)
#pragma unroll
        for (int nt = 0; nt < 2; nt++) acc[mt][nt] = vzero;

#if HAVE_TDM
    // ---------------- TDM double-buffered pipeline ----------------
    const int nkt = K / 32;
    const unsigned short* agb = A  + (size_t)rowBase * K;
    const unsigned short* bgb = Bt + (size_t)colBase * K;
    if (wave == 0) {
        tdm_load_tile((unsigned)(uintptr_t)&As[0][0], agb, K);
        tdm_load_tile((unsigned)(uintptr_t)&Bs[0][0], bgb, K);
    }
    for (int t = 0; t < nkt; t++) {
        const int buf = t & 1;
        __syncthreads();                 // prior readers of As[buf^1] done
        if (wave == 0) {
            if (t + 1 < nkt) {
                tdm_load_tile((unsigned)(uintptr_t)&As[buf ^ 1][0], agb + (t + 1) * 32, K);
                tdm_load_tile((unsigned)(uintptr_t)&Bs[buf ^ 1][0], bgb + (t + 1) * 32, K);
                __builtin_amdgcn_s_wait_tensorcnt(2);  // tile t resident
            } else {
                __builtin_amdgcn_s_wait_tensorcnt(0);
            }
        }
        __syncthreads();                 // data for tile t visible to all waves

        const unsigned short* ars = &As[buf][(waveM * 64 + m16) * 40 + hseg * 8];
        const unsigned short* brs = &Bs[buf][(waveN * 32 + m16) * 40 + hseg * 8];
        v16bf af[4], bfr[2];
#pragma unroll
        for (int mt = 0; mt < 4; mt++) {
            const unsigned short* p = ars + mt * 16 * 40;
            af[mt] = cat8(*reinterpret_cast<const v8bf*>(p),
                          *reinterpret_cast<const v8bf*>(p + 16));
        }
#pragma unroll
        for (int nt = 0; nt < 2; nt++) {
            const unsigned short* p = brs + nt * 16 * 40;
            bfr[nt] = cat8(*reinterpret_cast<const v8bf*>(p),
                           *reinterpret_cast<const v8bf*>(p + 16));
        }
#pragma unroll
        for (int mt = 0; mt < 4; mt++)
#pragma unroll
            for (int nt = 0; nt < 2; nt++)
                acc[mt][nt] = __builtin_amdgcn_wmma_f32_16x16x32_bf16(
                    false, af[mt], false, bfr[nt], (short)0, acc[mt][nt], false, false);
    }
#else
    // ---------------- fallback: manual global->LDS staging ----------------
    const int ldr = tid >> 1, seg = (tid & 1) * 16;
    const unsigned short* ag = A  + (size_t)(rowBase + ldr) * K + seg;
    const unsigned short* bg = Bt + (size_t)(colBase + ldr) * K + seg;
    unsigned short* asd = &As[0][ldr * 40 + seg];
    unsigned short* bsd = &Bs[0][ldr * 40 + seg];
    const unsigned short* ars = &As[0][(waveM * 64 + m16) * 40 + hseg * 8];
    const unsigned short* brs = &Bs[0][(waveN * 32 + m16) * 40 + hseg * 8];

    for (int kt = 0; kt < K; kt += 32) {
        uint4 a0 = *reinterpret_cast<const uint4*>(ag + kt);
        uint4 a1 = *reinterpret_cast<const uint4*>(ag + kt + 8);
        uint4 b0 = *reinterpret_cast<const uint4*>(bg + kt);
        uint4 b1 = *reinterpret_cast<const uint4*>(bg + kt + 8);
        if (kt + 32 < K) {
            __builtin_prefetch(ag + kt + 32, 0, 1);
            __builtin_prefetch(bg + kt + 32, 0, 1);
        }
        __syncthreads();
        *reinterpret_cast<uint4*>(asd)     = a0;
        *reinterpret_cast<uint4*>(asd + 8) = a1;
        *reinterpret_cast<uint4*>(bsd)     = b0;
        *reinterpret_cast<uint4*>(bsd + 8) = b1;
        __syncthreads();

        v16bf af[4], bfr[2];
#pragma unroll
        for (int mt = 0; mt < 4; mt++) {
            const unsigned short* p = ars + mt * 16 * 40;
            af[mt] = cat8(*reinterpret_cast<const v8bf*>(p),
                          *reinterpret_cast<const v8bf*>(p + 16));
        }
#pragma unroll
        for (int nt = 0; nt < 2; nt++) {
            const unsigned short* p = brs + nt * 16 * 40;
            bfr[nt] = cat8(*reinterpret_cast<const v8bf*>(p),
                           *reinterpret_cast<const v8bf*>(p + 16));
        }
#pragma unroll
        for (int mt = 0; mt < 4; mt++)
#pragma unroll
            for (int nt = 0; nt < 2; nt++)
                acc[mt][nt] = __builtin_amdgcn_wmma_f32_16x16x32_bf16(
                    false, af[mt], false, bfr[nt], (short)0, acc[mt][nt], false, false);
    }
#endif

    const float gc0 = 0.7978845608028654f, gc1 = 0.044715f;
#pragma unroll
    for (int mt = 0; mt < 4; mt++) {
#pragma unroll
        for (int nt = 0; nt < 2; nt++) {
            const int row0 = rowBase + waveM * 64 + mt * 16 + hseg * 8;
            const int col  = colBase + waveN * 32 + nt * 16 + m16;
#pragma unroll
            for (int r = 0; r < 8; r++) {
                const float vacc = acc[mt][nt][r];
                const size_t idx = (size_t)(row0 + r) * N + col;
                if (mode == 0) {
                    Cb[idx] = f2bf(vacc);
                } else if (mode == 1) {
                    float t = tanhf(gc0 * (vacc + gc1 * vacc * vacc * vacc));
                    Cb[idx] = f2bf(0.5f * vacc * (1.f + t));
                } else {
                    Cf[idx] = R[idx] + vacc;
                }
            }
        }
    }
}

// ---------------------------------------------------------------------------
// Causal flash attention: one wave (32 thr) per 16 query rows of one head.
// QK^T and PV via WMMA; online softmax with 16-lane shfl reductions.
// ---------------------------------------------------------------------------
__global__ void __launch_bounds__(32)
attn_kernel(const unsigned short* __restrict__ q,
            const unsigned short* __restrict__ k,
            const unsigned short* __restrict__ vt,
            unsigned short* __restrict__ ao) {
    __shared__ __align__(16) unsigned short Pt[16 * 40];
    const int lane = threadIdx.x;
    const int qt = blockIdx.x & (Sc / 16 - 1);        // 128 q-tiles
    const int h  = (blockIdx.x >> 7) & (NHc - 1);
    const int b0 = blockIdx.x >> 11;
    const int qBase = qt * 16;

    const unsigned short* Qh = q  + (size_t)b0 * Sc * Hc + (size_t)h * HDc;
    const unsigned short* Kh = k  + (size_t)b0 * Sc * Hc + (size_t)h * HDc;
    const unsigned short* Vh = vt + (size_t)(b0 * NHc + h) * HDc * Sc;

    const int m16 = lane & 15, hseg = lane >> 4;

    v16bf qf[4];                                      // Q 16x128 as 4 A-frags
#pragma unroll
    for (int c = 0; c < 4; c++) {
        const unsigned short* p = Qh + (size_t)(qBase + m16) * Hc + c * 32 + hseg * 8;
        qf[c] = cat8(*reinterpret_cast<const v8bf*>(p),
                     *reinterpret_cast<const v8bf*>(p + 16));
    }

    const v8f vzero = {0.f,0.f,0.f,0.f,0.f,0.f,0.f,0.f};
    v8f o[8];
#pragma unroll
    for (int dt = 0; dt < 8; dt++) o[dt] = vzero;
    float mrow[8], lrow[8];
#pragma unroll
    for (int r = 0; r < 8; r++) { mrow[r] = -1e30f; lrow[r] = 0.f; }

    const int nkc = (qBase + 16 + 31) >> 5;           // kv chunks of 32
    const float sc = 0.08838834764831845f;            // 1/sqrt(128)

    for (int kc = 0; kc < nkc; kc++) {
        const int kvBase = kc * 32;
        v8f s0 = vzero, s1 = vzero;
#pragma unroll
        for (int c = 0; c < 4; c++) {                 // reduce over d
            const unsigned short* p0 = Kh + (size_t)(kvBase + m16) * Hc + c * 32 + hseg * 8;
            v16bf kb0 = cat8(*reinterpret_cast<const v8bf*>(p0),
                             *reinterpret_cast<const v8bf*>(p0 + 16));
            s0 = __builtin_amdgcn_wmma_f32_16x16x32_bf16(false, qf[c], false, kb0,
                                                         (short)0, s0, false, false);
            const unsigned short* p1 = Kh + (size_t)(kvBase + 16 + m16) * Hc + c * 32 + hseg * 8;
            v16bf kb1 = cat8(*reinterpret_cast<const v8bf*>(p1),
                             *reinterpret_cast<const v8bf*>(p1 + 16));
            s1 = __builtin_amdgcn_wmma_f32_16x16x32_bf16(false, qf[c], false, kb1,
                                                         (short)0, s1, false, false);
        }
        const int col0 = kvBase + m16;
        const int rowb = qBase + hseg * 8;
#pragma unroll
        for (int r = 0; r < 8; r++) {
            const int row = rowb + r;
            float a0 = (col0      <= row) ? s0[r] * sc : -1e30f;
            float a1 = (col0 + 16 <= row) ? s1[r] * sc : -1e30f;
            float rm = fmaxf(a0, a1);
            rm = fmaxf(rm, __shfl_xor(rm, 1));
            rm = fmaxf(rm, __shfl_xor(rm, 2));
            rm = fmaxf(rm, __shfl_xor(rm, 4));
            rm = fmaxf(rm, __shfl_xor(rm, 8));        // within 16-lane half
            const float mnew = fmaxf(mrow[r], rm);
            const float corr = __expf(mrow[r] - mnew);
            const float p0e = __expf(a0 - mnew);
            const float p1e = __expf(a1 - mnew);
            float rs = p0e + p1e;
            rs += __shfl_xor(rs, 1);
            rs += __shfl_xor(rs, 2);
            rs += __shfl_xor(rs, 4);
            rs += __shfl_xor(rs, 8);
            lrow[r] = lrow[r] * corr + rs;
            mrow[r] = mnew;
#pragma unroll
            for (int dt = 0; dt < 8; dt++) o[dt][r] *= corr;
            Pt[(hseg * 8 + r) * 40 + m16]      = f2bf(p0e);
            Pt[(hseg * 8 + r) * 40 + 16 + m16] = f2bf(p1e);
        }
        __syncthreads();                              // C-layout -> A-layout via LDS
        const unsigned short* pp = Pt + m16 * 40 + hseg * 8;
        v16bf pf = cat8(*reinterpret_cast<const v8bf*>(pp),
                        *reinterpret_cast<const v8bf*>(pp + 16));
#pragma unroll
        for (int dt = 0; dt < 8; dt++) {
            const unsigned short* vp = Vh + (size_t)(dt * 16 + m16) * Sc + kvBase + hseg * 8;
            v16bf vf = cat8(*reinterpret_cast<const v8bf*>(vp),
                            *reinterpret_cast<const v8bf*>(vp + 16));
            o[dt] = __builtin_amdgcn_wmma_f32_16x16x32_bf16(false, pf, false, vf,
                                                            (short)0, o[dt], false, false);
        }
        __syncthreads();
    }

    unsigned short* aor = ao + (size_t)(b0 * Sc + qBase) * Hc + (size_t)h * HDc;
#pragma unroll
    for (int dt = 0; dt < 8; dt++)
#pragma unroll
        for (int r = 0; r < 8; r++) {
            const float val = o[dt][r] / lrow[r];
            aor[(size_t)(hseg * 8 + r) * Hc + dt * 16 + m16] = f2bf(val);
        }
}

// ---------------------------------------------------------------------------
extern "C" void kernel_launch(void* const* d_in, const int* in_sizes, int n_in,
                              void* d_out, int out_size, void* d_ws, size_t ws_size,
                              hipStream_t stream) {
    (void)in_sizes; (void)n_in; (void)out_size; (void)ws_size;
    const float* x     = (const float*)d_in[0];
    // d_in[1] = mask (causal, known statically) -- unused
    const float* wq    = (const float*)d_in[2];
    const float* wk    = (const float*)d_in[3];
    const float* wv    = (const float*)d_in[4];
    const float* wo    = (const float*)d_in[5];
    const float* w_in  = (const float*)d_in[6];
    const float* w_out = (const float*)d_in[7];
    const float* ln1g  = (const float*)d_in[8];
    const float* ln1b  = (const float*)d_in[9];
    const float* ln2g  = (const float*)d_in[10];
    const float* ln2b  = (const float*)d_in[11];
    float* out = (float*)d_out;

    char* ws = (char*)d_ws;
    const size_t SZ = (size_t)Bc * Sc * Hc * 2;       // bf16 bytes of one (B,S,H)
    unsigned short* xn    = (unsigned short*)(ws);             // reused as hn
    unsigned short* qb    = (unsigned short*)(ws + 1 * SZ);
    unsigned short* kb    = (unsigned short*)(ws + 2 * SZ);
    unsigned short* vb    = (unsigned short*)(ws + 3 * SZ);
    unsigned short* vtb   = (unsigned short*)(ws + 4 * SZ);
    unsigned short* aob   = (unsigned short*)(ws + 5 * SZ);
    unsigned short* wqT   = (unsigned short*)(ws + 6 * SZ);
    unsigned short* wkT   = (unsigned short*)(ws + 7 * SZ);
    unsigned short* wvT   = (unsigned short*)(ws + 8 * SZ);
    unsigned short* woT   = (unsigned short*)(ws + 9 * SZ);
    unsigned short* winT  = (unsigned short*)(ws + 10 * SZ);   // 2*SZ long
    unsigned short* woutT = (unsigned short*)(ws + 12 * SZ);   // 2*SZ long
    unsigned short* ff1   = qb;   // reuse q..vt region (exactly 4*SZ = 64MB)

    const dim3 tb(32, 8);
    const int Mrows = Bc * Sc;    // 4096

    // 1) weights -> bf16 transposed
    wcast_t_kernel<<<dim3(Hc / 32,  Hc / 32), tb, 0, stream>>>(wq,    wqT,   Hc,  Hc);
    wcast_t_kernel<<<dim3(Hc / 32,  Hc / 32), tb, 0, stream>>>(wk,    wkT,   Hc,  Hc);
    wcast_t_kernel<<<dim3(Hc / 32,  Hc / 32), tb, 0, stream>>>(wv,    wvT,   Hc,  Hc);
    wcast_t_kernel<<<dim3(Hc / 32,  Hc / 32), tb, 0, stream>>>(wo,    woT,   Hc,  Hc);
    wcast_t_kernel<<<dim3(FFc / 32, Hc / 32), tb, 0, stream>>>(w_in,  winT,  Hc,  FFc);
    wcast_t_kernel<<<dim3(Hc / 32, FFc / 32), tb, 0, stream>>>(w_out, woutT, FFc, Hc);

    // 2) pre-norm + QKV projections
    ln_bf16_kernel<<<Bc * Sc, 256, 0, stream>>>(x, ln1g, ln1b, xn);
    gemm_bf16_kernel<<<dim3(Hc / 128, Mrows / 128), 256, 0, stream>>>(xn, wqT, qb, nullptr, nullptr, Mrows, Hc, Hc, 0);
    gemm_bf16_kernel<<<dim3(Hc / 128, Mrows / 128), 256, 0, stream>>>(xn, wkT, kb, nullptr, nullptr, Mrows, Hc, Hc, 0);
    gemm_bf16_kernel<<<dim3(Hc / 128, Mrows / 128), 256, 0, stream>>>(xn, wvT, vb, nullptr, nullptr, Mrows, Hc, Hc, 0);

    // 3) attention
    vtrans_kernel<<<dim3(Sc / 32, HDc / 32, Bc * NHc), tb, 0, stream>>>(vb, vtb);
    attn_kernel<<<Bc * NHc * (Sc / 16), 32, 0, stream>>>(qb, kb, vtb, aob);

    // 4) h = x + ao @ wo   (f32 into d_out)
    gemm_bf16_kernel<<<dim3(Hc / 128, Mrows / 128), 256, 0, stream>>>(aob, woT, nullptr, out, x, Mrows, Hc, Hc, 2);

    // 5) FFN with pre-norm + GELU, residual into d_out
    ln_bf16_kernel<<<Bc * Sc, 256, 0, stream>>>(out, ln2g, ln2b, xn);
    gemm_bf16_kernel<<<dim3(FFc / 128, Mrows / 128), 256, 0, stream>>>(xn, winT, ff1, nullptr, nullptr, Mrows, FFc, Hc, 1);
    gemm_bf16_kernel<<<dim3(Hc / 128, Mrows / 128), 256, 0, stream>>>(ff1, woutT, nullptr, out, out, Mrows, Hc, FFc, 2);
}
